// MultiheadAttention_1219770712615
// MI455X (gfx1250) — compile-verified, matrix operand order checked by reading, not run on hardware
//
#include <hip/hip_runtime.h>

// ---------------------------------------------------------------------------
// MI455X (gfx1250, wave32) fused multihead attention:
//   proj(Q,K,V) [bf16 WMMA] -> flash attention [bf16 WMMA, online softmax,
//   async global->LDS staging] -> out projection [bf16 WMMA].
// GEMMs: block tile 128x32, BK=64, 8 waves, 4 WMMAs/wave/stage.
// ---------------------------------------------------------------------------

typedef __attribute__((ext_vector_type(16))) __bf16 v16bf;
typedef __attribute__((ext_vector_type(8)))  __bf16 v8bf;
typedef __attribute__((ext_vector_type(4)))  __bf16 v4bf;
typedef __attribute__((ext_vector_type(8)))  float  v8f;

#define WMMA_BF16(a, b, c) \
  __builtin_amdgcn_wmma_f32_16x16x32_bf16(false, (a), false, (b), (short)0, (c), false, false)

constexpr int kBS  = 4;
constexpr int kSEQ = 2048;
constexpr int kDIM = 1024;
constexpr int kNH  = 16;
constexpr int kDPH = 64;
constexpr int kM   = kBS * kSEQ;  // 8192 rows

__device__ __forceinline__ v8f zero8() {
  v8f z = {0.f, 0.f, 0.f, 0.f, 0.f, 0.f, 0.f, 0.f};
  return z;
}

// Async 16-byte global->LDS copy (GLOBAL_LOAD_ASYNC_TO_LDS_B128, ASYNCcnt).
// Generic pointers to __shared__ objects carry the LDS byte address in their
// low 32 bits (ISA 10.2 aperture mapping), which is
// exactly what the VDST operand of the async load expects.
__device__ __forceinline__ void async_copy_b128(void* lds_dst, const void* gsrc) {
  unsigned           loff = (unsigned)(unsigned long long)lds_dst;
  unsigned long long ga   = (unsigned long long)gsrc;
  asm volatile("global_load_async_to_lds_b128 %0, %1, off"
               :: "v"(loff), "v"(ga) : "memory");
}

__device__ __forceinline__ void wait_async() {
  asm volatile("s_wait_asynccnt 0x0" ::: "memory");
}

// Load a 16x16x32 bf16 A/B fragment (16 elements / lane) from a row-major LDS
// row pointer.  ISA layout (CDNA5 ISA):
//   lanes 0-15 : elements 0..7 = K 0..7,  elements 8..15 = K 16..23
//   lanes 16-31: elements 0..7 = K 8..15, elements 8..15 = K 24..31
// Two contiguous 8-element runs -> ds_load_b128 pairs (rows are 16B-aligned).
__device__ __forceinline__ v16bf load_frag(const __bf16* row, int lane) {
  v16bf f;
  const int kb = (lane & 16) ? 8 : 0;
#pragma unroll
  for (int i = 0; i < 8; ++i) {
    f[i]     = row[kb + i];
    f[i + 8] = row[kb + 16 + i];
  }
  return f;
}

__device__ __forceinline__ v4bf pack4(float4 f) {
  v4bf p;
  p[0] = (__bf16)f.x; p[1] = (__bf16)f.y;
  p[2] = (__bf16)f.z; p[3] = (__bf16)f.w;
  return p;
}

// ---------------------------------------------------------------------------
// Projection GEMM:  out = (A[M,1024] @ W[1024,1024]^T + bias) * scale
// scattered into [B][NH][SEQ][DPH] bf16 (head-major) for the attention stage.
// Block = 256 threads = 8 waves.  Tile M=128, N=32, K=64.
// Wave wv owns rows [wv*16, wv*16+16) x all 32 cols: 2 accumulators,
// A-fragments reused across the two N tiles -> 4 WMMAs / wave / stage.
// ---------------------------------------------------------------------------
__global__ __launch_bounds__(256) void proj_gemm_kernel(
    const float* __restrict__ A, const float* __restrict__ W,
    const float* __restrict__ bias, __bf16* __restrict__ out, float scale) {
  constexpr int LDT = 80;  // row stride (bf16): 64 + 16 pad, 16B-aligned rows
  __shared__ __align__(16) __bf16 As[128 * LDT];  // 20.0 KB
  __shared__ __align__(16) __bf16 Ws[32 * LDT];   //  5.0 KB

  const int tid  = threadIdx.x;
  const int lane = tid & 31;
  const int wv   = tid >> 5;
  const int l15  = lane & 15;
  const int m0   = blockIdx.y * 128;
  const int n0   = blockIdx.x * 32;

  const float4* A4 = (const float4*)A;
  const float4* W4 = (const float4*)W;

  v8f acc[2] = {zero8(), zero8()};

  for (int kc = 0; kc < kDIM; kc += 64) {
    __syncthreads();
    // A tile: 128 rows x 64 cols = 2048 float4, 8 per thread.
#pragma unroll
    for (int i = 0; i < 8; ++i) {
      int e = tid + i * 256;
      int r = e >> 4, c4 = e & 15;
      float4 f = A4[(size_t)(m0 + r) * (kDIM / 4) + (kc >> 2) + c4];
      *(v4bf*)&As[r * LDT + c4 * 4] = pack4(f);
    }
    // W tile: 32 rows x 64 cols = 512 float4, 2 per thread.
#pragma unroll
    for (int i = 0; i < 2; ++i) {
      int e = tid + i * 256;
      int r = e >> 4, c4 = e & 15;
      float4 f = W4[(size_t)(n0 + r) * (kDIM / 4) + (kc >> 2) + c4];
      *(v4bf*)&Ws[r * LDT + c4 * 4] = pack4(f);
    }
    __syncthreads();

    const __bf16* arow = &As[(wv * 16 + l15) * LDT];
    v16bf a0 = load_frag(arow + 0, lane);
    v16bf a1 = load_frag(arow + 32, lane);
#pragma unroll
    for (int n = 0; n < 2; ++n) {
      const __bf16* brow = &Ws[(n * 16 + l15) * LDT];
      v16bf b0 = load_frag(brow + 0, lane);
      acc[n] = WMMA_BF16(a0, b0, acc[n]);
      v16bf b1 = load_frag(brow + 32, lane);
      acc[n] = WMMA_BF16(a1, b1, acc[n]);
    }
  }

  const int roff = (lane & 16) ? 8 : 0;
#pragma unroll
  for (int n = 0; n < 2; ++n) {
    const int   nl = n0 + n * 16 + l15;
    const float bv = bias[nl];
    const int   h  = nl >> 6;   // head
    const int   d  = nl & 63;   // dim-per-head
#pragma unroll
    for (int r = 0; r < 8; ++r) {
      int   m  = m0 + wv * 16 + r + roff;
      int   bb = m >> 11;         // batch
      int   s  = m & (kSEQ - 1);  // seq position
      float vv = (acc[n][r] + bv) * scale;
      out[(((size_t)bb * kNH + h) * kSEQ + s) * kDPH + d] = (__bf16)vv;
    }
  }
}

// ---------------------------------------------------------------------------
// Output GEMM:  out[M,1024] = ctx_bf16[M,1024] @ Wo^T + bo   (f32 output)
// Same tiling as proj_gemm; A is already bf16 -> 16B copies into LDS.
// ---------------------------------------------------------------------------
__global__ __launch_bounds__(256) void out_gemm_kernel(
    const __bf16* __restrict__ A, const float* __restrict__ W,
    const float* __restrict__ bias, float* __restrict__ out) {
  constexpr int LDT = 80;
  __shared__ __align__(16) __bf16 As[128 * LDT];
  __shared__ __align__(16) __bf16 Ws[32 * LDT];

  const int tid  = threadIdx.x;
  const int lane = tid & 31;
  const int wv   = tid >> 5;
  const int l15  = lane & 15;
  const int m0   = blockIdx.y * 128;
  const int n0   = blockIdx.x * 32;

  const uint4*  A16 = (const uint4*)A;  // 8 bf16 per uint4
  const float4* W4  = (const float4*)W;

  v8f acc[2] = {zero8(), zero8()};

  for (int kc = 0; kc < kDIM; kc += 64) {
    __syncthreads();
    // A tile: 128 rows x 64 bf16 = 1024 x 16B, 4 per thread, async to LDS.
#pragma unroll
    for (int i = 0; i < 4; ++i) {
      int e = tid + i * 256;
      int r = e >> 3, c8 = e & 7;
      async_copy_b128(&As[r * LDT + c8 * 8],
                      &A16[(size_t)(m0 + r) * (kDIM / 8) + (kc >> 3) + c8]);
    }
#pragma unroll
    for (int i = 0; i < 2; ++i) {
      int e = tid + i * 256;
      int r = e >> 4, c4 = e & 15;
      float4 f = W4[(size_t)(n0 + r) * (kDIM / 4) + (kc >> 2) + c4];
      *(v4bf*)&Ws[r * LDT + c4 * 4] = pack4(f);
    }
    wait_async();
    __syncthreads();

    const __bf16* arow = &As[(wv * 16 + l15) * LDT];
    v16bf a0 = load_frag(arow + 0, lane);
    v16bf a1 = load_frag(arow + 32, lane);
#pragma unroll
    for (int n = 0; n < 2; ++n) {
      const __bf16* brow = &Ws[(n * 16 + l15) * LDT];
      v16bf b0 = load_frag(brow + 0, lane);
      acc[n] = WMMA_BF16(a0, b0, acc[n]);
      v16bf b1 = load_frag(brow + 32, lane);
      acc[n] = WMMA_BF16(a1, b1, acc[n]);
    }
  }

  const int roff = (lane & 16) ? 8 : 0;
#pragma unroll
  for (int n = 0; n < 2; ++n) {
    const int   nl = n0 + n * 16 + l15;
    const float bv = bias[nl];
#pragma unroll
    for (int r = 0; r < 8; ++r) {
      int m = m0 + wv * 16 + r + roff;
      out[(size_t)m * kDIM + nl] = acc[n][r] + bv;
    }
  }
}

// ---------------------------------------------------------------------------
// Flash attention.  Grid: (SEQ/128, BS*NH).  Block = 256 threads = 8 waves,
// each wave owns 16 query rows.  Iterates 64-key tiles with online softmax.
// Q/K tiles staged with GLOBAL_LOAD_ASYNC_TO_LDS_B128 (ASYNCcnt).
// Q is pre-scaled by 1/sqrt(DPH) in the projection.
// ---------------------------------------------------------------------------
__global__ __launch_bounds__(256) void flash_attn_kernel(
    const __bf16* __restrict__ q, const __bf16* __restrict__ k,
    const __bf16* __restrict__ v, const unsigned char* __restrict__ mask,
    __bf16* __restrict__ ctx) {
  constexpr int TS = 80;  // row stride (bf16): 16B-aligned rows, bank spread
  __shared__ __align__(16) __bf16 Qs[128 * TS];     // 128 q-rows x 64 dph
  __shared__ __align__(16) __bf16 Ks[64 * TS];      // 64 keys  x 64 dph
  __shared__ __align__(16) __bf16 Vt[64 * TS];      // 64 dph x 64 keys (V^T)
  __shared__ __align__(16) __bf16 Ps[8 * 16 * TS];  // per-wave P staging

  const int tid  = threadIdx.x;
  const int lane = tid & 31;
  const int wv   = tid >> 5;
  const int l15  = lane & 15;
  const int bh   = blockIdx.y;  // b*NH + h
  const int bidx = bh >> 4;
  const int h    = bh & 15;
  const int q0   = blockIdx.x * 128;

  const __bf16* qb = q + (size_t)bh * kSEQ * kDPH;
  const __bf16* kb = k + (size_t)bh * kSEQ * kDPH;
  const __bf16* vb = v + (size_t)bh * kSEQ * kDPH;
  const unsigned char* mb = mask + (size_t)bidx * kSEQ;

  // Q tile: 128 rows x 64 bf16 = 1024 x 16B, 4 per thread, async to LDS.
  const uint4* q16 = (const uint4*)(qb + (size_t)q0 * kDPH);
#pragma unroll
  for (int i = 0; i < 4; ++i) {
    int e = tid + i * 256;
    int r = e >> 3, c8 = e & 7;
    async_copy_b128(&Qs[r * TS + c8 * 8], &q16[r * 8 + c8]);
  }
  wait_async();
  __syncthreads();

  const v16bf aq0 = load_frag(&Qs[(wv * 16 + l15) * TS + 0], lane);
  const v16bf aq1 = load_frag(&Qs[(wv * 16 + l15) * TS + 32], lane);

  float mx[8], lsum[8];
  v8f   accO[4];
#pragma unroll
  for (int r = 0; r < 8; ++r) { mx[r] = -3.0e38f; lsum[r] = 0.f; }
#pragma unroll
  for (int j = 0; j < 4; ++j) accO[j] = zero8();

  for (int kt = 0; kt < kSEQ / 64; ++kt) {
    const int ks0 = kt * 64;
    __syncthreads();
    // K tile: 64 x 64 bf16 = 512 x 16B, 2 per thread, async to LDS.
    {
      const uint4* k16 = (const uint4*)(kb + (size_t)ks0 * kDPH);
#pragma unroll
      for (int i = 0; i < 2; ++i) {
        int e = tid + i * 256;
        int r = e >> 3, c8 = e & 7;
        async_copy_b128(&Ks[r * TS + c8 * 8], &k16[r * 8 + c8]);
      }
      // V tile, transposed: read 8 bf16 (b128), scatter b16 stores.
      const v8bf* v8p = (const v8bf*)(vb + (size_t)ks0 * kDPH);
#pragma unroll
      for (int i = 0; i < 2; ++i) {
        int e = tid + i * 256;
        int r = e >> 3, c8 = e & 7;
        v8bf vv = v8p[r * 8 + c8];
#pragma unroll
        for (int x = 0; x < 8; ++x) Vt[(c8 * 8 + x) * TS + r] = vv[x];
      }
    }
    if (kt + 1 < kSEQ / 64) {  // global_prefetch_b8 for next K/V tiles
      __builtin_prefetch(kb + (size_t)(ks0 + 64) * kDPH + tid * 16, 0, 1);
      __builtin_prefetch(vb + (size_t)(ks0 + 64) * kDPH + tid * 16, 0, 1);
    }
    wait_async();
    __syncthreads();

    // S = Q . K^T  (16 rows x 64 keys per wave), 4 col-tiles x 2 K-chunks
    v8f sacc[4];
#pragma unroll
    for (int j = 0; j < 4; ++j) {
      sacc[j] = zero8();
      v16bf b0 = load_frag(&Ks[(j * 16 + l15) * TS + 0], lane);
      sacc[j] = WMMA_BF16(aq0, b0, sacc[j]);
      v16bf b1 = load_frag(&Ks[(j * 16 + l15) * TS + 32], lane);
      sacc[j] = WMMA_BF16(aq1, b1, sacc[j]);
      if (!mb[ks0 + j * 16 + l15]) {  // key padding mask (True = valid)
#pragma unroll
        for (int r = 0; r < 8; ++r) sacc[j][r] = -3.0e38f;
      }
    }

    // Online softmax.  C-layout: lane&15 = column, row = r + 8*(lane>=16),
    // so row reductions are across the 16-lane halves (width-16 shuffles).
    float rmax[8];
#pragma unroll
    for (int r = 0; r < 8; ++r)
      rmax[r] = fmaxf(fmaxf(sacc[0][r], sacc[1][r]), fmaxf(sacc[2][r], sacc[3][r]));
#pragma unroll
    for (int msk = 1; msk < 16; msk <<= 1)
#pragma unroll
      for (int r = 0; r < 8; ++r)
        rmax[r] = fmaxf(rmax[r], __shfl_xor(rmax[r], msk, 16));

#pragma unroll
    for (int r = 0; r < 8; ++r) {
      float mn = fmaxf(mx[r], rmax[r]);
      float al = (mx[r] > -1.0e37f) ? __expf(mx[r] - mn) : 0.0f;
      mx[r] = mn;
      lsum[r] *= al;
#pragma unroll
      for (int j = 0; j < 4; ++j) accO[j][r] *= al;
    }

    float rsum[8];
#pragma unroll
    for (int r = 0; r < 8; ++r) rsum[r] = 0.f;
#pragma unroll
    for (int j = 0; j < 4; ++j)
#pragma unroll
      for (int r = 0; r < 8; ++r) {
        float p = __expf(sacc[j][r] - mx[r]);
        sacc[j][r] = p;
        rsum[r] += p;
      }
#pragma unroll
    for (int msk = 1; msk < 16; msk <<= 1)
#pragma unroll
      for (int r = 0; r < 8; ++r) rsum[r] += __shfl_xor(rsum[r], msk, 16);
#pragma unroll
    for (int r = 0; r < 8; ++r) lsum[r] += rsum[r];

    // Stage P (bf16) into this wave's private LDS region: converts the
    // C-layout score registers into A-fragment layout for the PV matmul.
    {
      const int roff = (lane & 16) ? 8 : 0;
      __bf16* pw = &Ps[(size_t)wv * 16 * TS];
#pragma unroll
      for (int j = 0; j < 4; ++j)
#pragma unroll
        for (int r = 0; r < 8; ++r)
          pw[(r + roff) * TS + j * 16 + l15] = (__bf16)sacc[j][r];

      // Same-wave LDS ops are in-order: stores above complete before loads.
      v16bf ap0 = load_frag(&pw[l15 * TS + 0], lane);
      v16bf ap1 = load_frag(&pw[l15 * TS + 32], lane);
#pragma unroll
      for (int j = 0; j < 4; ++j) {
        v16bf b0 = load_frag(&Vt[(j * 16 + l15) * TS + 0], lane);
        accO[j] = WMMA_BF16(ap0, b0, accO[j]);
        v16bf b1 = load_frag(&Vt[(j * 16 + l15) * TS + 32], lane);
        accO[j] = WMMA_BF16(ap1, b1, accO[j]);
      }
    }
  }

  // Normalize and emit context as [B][SEQ][DIM] bf16 (row-major for out GEMM).
  const int roff = (lane & 16) ? 8 : 0;
#pragma unroll
  for (int r = 0; r < 8; ++r) {
    float inv  = (lsum[r] > 0.f) ? 1.0f / lsum[r] : 0.f;
    int   srow = q0 + wv * 16 + r + roff;
#pragma unroll
    for (int j = 0; j < 4; ++j) {
      int d = j * 16 + l15;
      ctx[((size_t)bidx * kSEQ + srow) * kDIM + h * kDPH + d] =
          (__bf16)(accO[j][r] * inv);
    }
  }
}

// ---------------------------------------------------------------------------
// Launch: 3 projections -> flash attention -> output projection (one stream,
// sequential dependencies, graph-capture safe; workspace = 67 MB bf16).
// ---------------------------------------------------------------------------
extern "C" void kernel_launch(void* const* d_in, const int* in_sizes, int n_in,
                              void* d_out, int out_size, void* d_ws, size_t ws_size,
                              hipStream_t stream) {
  const float* query = (const float*)d_in[0];
  const float* key_  = (const float*)d_in[1];
  const float* value = (const float*)d_in[2];
  const unsigned char* mask = (const unsigned char*)d_in[3];
  const float* Wq = (const float*)d_in[4];
  const float* bq = (const float*)d_in[5];
  const float* Wk = (const float*)d_in[6];
  const float* bk = (const float*)d_in[7];
  const float* Wv = (const float*)d_in[8];
  const float* bv = (const float*)d_in[9];
  const float* Wo = (const float*)d_in[10];
  const float* bo = (const float*)d_in[11];

  const size_t qkv = (size_t)kM * kDIM;  // 8,388,608 bf16 elements each
  __bf16* qw = (__bf16*)d_ws;
  __bf16* kw = qw + qkv;
  __bf16* vw = kw + qkv;
  __bf16* cw = vw + qkv;

  dim3 blk(256);
  dim3 gemm_grid(kDIM / 32, kM / 128);  // (32, 64)

  hipLaunchKernelGGL(proj_gemm_kernel, gemm_grid, blk, 0, stream,
                     query, Wq, bq, qw, 0.125f);   // fused 1/sqrt(DPH)
  hipLaunchKernelGGL(proj_gemm_kernel, gemm_grid, blk, 0, stream,
                     key_, Wk, bk, kw, 1.0f);
  hipLaunchKernelGGL(proj_gemm_kernel, gemm_grid, blk, 0, stream,
                     value, Wv, bv, vw, 1.0f);

  dim3 attn_grid(kSEQ / 128, kBS * kNH);  // (16, 64)
  hipLaunchKernelGGL(flash_attn_kernel, attn_grid, blk, 0, stream,
                     qw, kw, vw, mask, cw);

  hipLaunchKernelGGL(out_gemm_kernel, gemm_grid, blk, 0, stream,
                     cw, Wo, bo, (float*)d_out);

  (void)in_sizes; (void)n_in; (void)out_size; (void)ws_size;
}
